// GRU_DEC_ATTN_17205638987916
// MI455X (gfx1250) — matrix operand order repaired and checked by reading, not compiled, not run on hardware
//
#include <hip/hip_runtime.h>
#include <hip/hip_bf16.h>

typedef _Float16 v8h  __attribute__((ext_vector_type(8)));
typedef _Float16 v16h __attribute__((ext_vector_type(16)));
typedef float    v8f  __attribute__((ext_vector_type(8)));
typedef int      v4i  __attribute__((vector_size(16)));   // matches builtin's int4

#define Bsz 4096
#define Hsz 1024
#define INsz 784
#define INpad 800
#define Lsz 64
#define A_STRIDE 40   // 32 halfs + pad (80B row, breaks LDS bank conflicts, keeps 16B align)

// ---------------- async global -> LDS staging (CDNA5 path with fallback) ----------------
#if __has_builtin(__builtin_amdgcn_global_load_async_to_lds_b128)
#define HAVE_ASYNC 1
__device__ __forceinline__ void cp16(const _Float16* g, _Float16* l) {
    __builtin_amdgcn_global_load_async_to_lds_b128(
        (__attribute__((address_space(1))) v4i*)(g),
        (__attribute__((address_space(3))) v4i*)(l), 0, 0);
}
__device__ __forceinline__ void wait_async() {
    asm volatile("s_wait_asynccnt 0x0" ::: "memory");
}
#else
#define HAVE_ASYNC 0
__device__ __forceinline__ void cp16(const _Float16* g, _Float16* l) {
    *(v8h*)l = *(const v8h*)g;   // sync fallback; barrier handles visibility
}
__device__ __forceinline__ void wait_async() {}
#endif

// stage one 128x32(K) A tile and 128x32(K) B tile; 256 threads x 4 b128 copies
__device__ __forceinline__ void stage(
    _Float16* __restrict__ dA, _Float16* __restrict__ dB,
    const _Float16* __restrict__ A1, const _Float16* __restrict__ A2,
    int Ksplit, int Kdim, const _Float16* __restrict__ bRow,
    int blockM, int sr, int sc, int k0)
{
    const _Float16* aRow = (k0 < Ksplit)
        ? (A1 + (size_t)(blockM + sr) * (size_t)Ksplit + (size_t)(k0 + sc))
        : (A2 + (size_t)(blockM + sr) * (size_t)(Kdim - Ksplit) + (size_t)(k0 - Ksplit + sc));
    _Float16* la = dA + sr * A_STRIDE + sc;
    cp16(aRow,     la);
    cp16(aRow + 8, la + 8);
    const _Float16* b = bRow + k0;
    _Float16* lb = dB + sr * A_STRIDE + sc;
    cp16(b,     lb);
    cp16(b + 8, lb + 8);
}

// C[M,N] = act( A[M,K] @ W[N,K]^T + bias ); A optionally (A1|A2) split at Ksplit.
// All operands f16, K % 32 == 0. Block tile 128x128, 8 waves, wave tile 32x64 (2x4 wmma).
// Double-buffered LDS with async global->LDS copies.
template<int ACT, int WF32, int WF16>
__global__ __launch_bounds__(256)
void wmma_gemm(const _Float16* __restrict__ A1, const _Float16* __restrict__ A2,
               int Ksplit, int Kdim,
               const _Float16* __restrict__ W, const float* __restrict__ bias,
               float* __restrict__ C32, _Float16* __restrict__ C16, int Ndim)
{
    __shared__ _Float16 sA[2][128 * A_STRIDE];
    __shared__ _Float16 sB[2][128 * A_STRIDE];

    const int tid   = threadIdx.x;
    const int lane  = tid & 31;
    const int wid   = tid >> 5;
    const int waveM = wid >> 1;                 // 0..3 -> 32 rows each
    const int waveN = wid & 1;                  // 0..1 -> 64 cols each
    const int blockM = blockIdx.y * 128;
    const int blockN = blockIdx.x * 128;

    const int halfSel = (lane < 16) ? 0 : 1;
    const int nrow    = lane & 15;

    // staging map: 2 threads per row, 16 halfs (2 x b128) each, same for A and B
    const int sr = tid >> 1;
    const int sc = (tid & 1) * 16;

    // B staging row pointer (clamp OOB rows to a valid one; epilogue masks stores)
    int gnB = blockN + sr; if (gnB > Ndim - 1) gnB = Ndim - 1;
    const _Float16* bRow = W + (size_t)gnB * Kdim + sc;

    v8f acc[2][4];
    #pragma unroll
    for (int i = 0; i < 2; ++i)
        #pragma unroll
        for (int j = 0; j < 4; ++j)
            acc[i][j] = (v8f)0.f;

    // prologue: stage k0 = 0 into buffer 0
    stage(sA[0], sB[0], A1, A2, Ksplit, Kdim, bRow, blockM, sr, sc, 0);

    int cur = 0;
    for (int k0 = 0; k0 < Kdim; k0 += 32) {
        wait_async();        // this wave's copies into sX[cur] complete
        __syncthreads();     // all waves' copies complete / prior reads done

        if (k0 + 32 < Kdim)  // prefetch next tile while computing this one
            stage(sA[cur ^ 1], sB[cur ^ 1], A1, A2, Ksplit, Kdim, bRow, blockM, sr, sc, k0 + 32);

        // fragments (ISA 7.12.2 16-bit layouts)
        v16h aF[2], bF[4];
        #pragma unroll
        for (int tm = 0; tm < 2; ++tm) {
            const int r = waveM * 32 + tm * 16 + nrow;
            const _Float16* p = &sA[cur][r * A_STRIDE + halfSel * 8];
            v8h lo = *(const v8h*)(p);
            v8h hi = *(const v8h*)(p + 16);
            aF[tm] = __builtin_shufflevector(lo, hi, 0,1,2,3,4,5,6,7,8,9,10,11,12,13,14,15);
        }
        #pragma unroll
        for (int tn = 0; tn < 4; ++tn) {
            const int n = waveN * 64 + tn * 16 + nrow;
            const _Float16* p = &sB[cur][n * A_STRIDE + halfSel * 8];
            v8h lo = *(const v8h*)(p);
            v8h hi = *(const v8h*)(p + 16);
            bF[tn] = __builtin_shufflevector(lo, hi, 0,1,2,3,4,5,6,7,8,9,10,11,12,13,14,15);
        }
        #pragma unroll
        for (int tm = 0; tm < 2; ++tm)
            #pragma unroll
            for (int tn = 0; tn < 4; ++tn)
                acc[tm][tn] = __builtin_amdgcn_wmma_f32_16x16x32_f16(
                    false, aF[tm], false, bF[tn], (short)0, acc[tm][tn], false, false);

        cur ^= 1;
    }

    // epilogue: VGPR v -> M = v + (lane<16?0:8), N = lane&15
    const int mOff = halfSel * 8;
    #pragma unroll
    for (int tm = 0; tm < 2; ++tm) {
        #pragma unroll
        for (int tn = 0; tn < 4; ++tn) {
            const int gn = blockN + waveN * 64 + tn * 16 + nrow;
            if (gn < Ndim) {
                const float bv = bias[gn];
                #pragma unroll
                for (int v = 0; v < 8; ++v) {
                    const int gm = blockM + waveM * 32 + tm * 16 + mOff + v;
                    float val = acc[tm][tn][v] + bv;
                    if (ACT == 1) val = fmaxf(val, 0.f);
                    if (WF32) C32[(size_t)gm * Ndim + gn] = val;
                    if (WF16) C16[(size_t)gm * Ndim + gn] = (_Float16)val;
                }
            }
        }
    }
}

// fp32 -> f16 with optional column zero-padding (dcols % 8 == 0)
__global__ __launch_bounds__(256)
void cvt16(const float* __restrict__ src, _Float16* __restrict__ dst,
           int scols, int dcols, long total)
{
    const long idx = ((long)blockIdx.x * 256 + threadIdx.x) * 8;
    if (idx >= total) return;
    const int r = (int)(idx / dcols);
    const int c = (int)(idx - (long)r * dcols);
    const float* s = src + (size_t)r * scols + c;
    #pragma unroll
    for (int j = 0; j < 8; ++j)
        dst[idx + j] = (c + j < scols) ? (_Float16)s[j] : (_Float16)0.f;
}

// softmax over rows of 64, one wave per row
__global__ void softmax64(const float* __restrict__ in, float* __restrict__ out) {
    const int b = blockIdx.x, t = threadIdx.x;
    float v0 = in[(size_t)b * 64 + t];
    float v1 = in[(size_t)b * 64 + 32 + t];
    float m = fmaxf(v0, v1);
    #pragma unroll
    for (int off = 16; off > 0; off >>= 1) m = fmaxf(m, __shfl_xor(m, off, 32));
    float e0 = __expf(v0 - m), e1 = __expf(v1 - m);
    float s = e0 + e1;
    #pragma unroll
    for (int off = 16; off > 0; off >>= 1) s += __shfl_xor(s, off, 32);
    const float r = 1.f / s;
    out[(size_t)b * 64 + t]      = e0 * r;
    out[(size_t)b * 64 + 32 + t] = e1 * r;
}

// attn_applied[b,h] = sum_l w[b,l]*enc[b,l,h]; bandwidth-bound; writes f16 for next GEMM
__global__ __launch_bounds__(256)
void attn_apply(const float* __restrict__ w, const float* __restrict__ enc,
                _Float16* __restrict__ out16) {
    __shared__ float sw[64];
    const int b = blockIdx.x;
    const int h = blockIdx.y * 256 + threadIdx.x;
    if (threadIdx.x < 64) sw[threadIdx.x] = w[(size_t)b * 64 + threadIdx.x];
    __syncthreads();
    const float* e = enc + (size_t)b * 64 * Hsz + h;
    float acc = 0.f;
    #pragma unroll
    for (int l = 0; l < 64; ++l) acc += sw[l] * e[(size_t)l * Hsz];
    out16[(size_t)b * Hsz + h] = (_Float16)acc;
}

__global__ void gru_gate(const float* __restrict__ gi, const float* __restrict__ gh,
                         const float* __restrict__ h0,
                         float* __restrict__ hnew32, _Float16* __restrict__ hnew16) {
    const size_t i = (size_t)blockIdx.x * 256 + threadIdx.x;   // over B*H
    const size_t b = i >> 10;
    const int    h = (int)(i & 1023);
    const size_t base = b * 3072;
    const float gir = gi[base + h], giz = gi[base + 1024 + h], gin = gi[base + 2048 + h];
    const float ghr = gh[base + h], ghz = gh[base + 1024 + h], ghn = gh[base + 2048 + h];
    const float r = 1.f / (1.f + __expf(-(gir + ghr)));
    const float z = 1.f / (1.f + __expf(-(giz + ghz)));
    const float n = tanhf(gin + r * ghn);
    const float hv = (1.f - z) * n + z * h0[i];
    hnew32[i] = hv;
    hnew16[i] = (_Float16)hv;
}

// in-place log_softmax over rows of 784
__global__ __launch_bounds__(256)
void logsoftmax784(float* __restrict__ x) {
    __shared__ float red[8];
    const int tid = threadIdx.x;
    float* row = x + (size_t)blockIdx.x * 784;
    float m = -3.0e38f;
    for (int k = tid; k < 784; k += 256) m = fmaxf(m, row[k]);
    #pragma unroll
    for (int off = 16; off > 0; off >>= 1) m = fmaxf(m, __shfl_xor(m, off, 32));
    if ((tid & 31) == 0) red[tid >> 5] = m;
    __syncthreads();
    float mb = red[0];
    #pragma unroll
    for (int i = 1; i < 8; ++i) mb = fmaxf(mb, red[i]);
    __syncthreads();
    float s = 0.f;
    for (int k = tid; k < 784; k += 256) s += __expf(row[k] - mb);
    #pragma unroll
    for (int off = 16; off > 0; off >>= 1) s += __shfl_xor(s, off, 32);
    if ((tid & 31) == 0) red[tid >> 5] = s;
    __syncthreads();
    float sb = 0.f;
    #pragma unroll
    for (int i = 0; i < 8; ++i) sb += red[i];
    const float lse = mb + __logf(sb);
    for (int k = tid; k < 784; k += 256) row[k] -= lse;
}

static inline int cvtBlocks(long total) { return (int)((total / 8 + 255) / 256); }

extern "C" void kernel_launch(void* const* d_in, const int* in_sizes, int n_in,
                              void* d_out, int out_size, void* d_ws, size_t ws_size,
                              hipStream_t stream) {
    const float* input  = (const float*)d_in[0];
    const float* hidden = (const float*)d_in[1];   // (1,B,H)
    const float* enc    = (const float*)d_in[2];   // (B,L,1,H)
    const float* W_emb  = (const float*)d_in[3];
    const float* b_emb  = (const float*)d_in[4];
    const float* W_attn = (const float*)d_in[5];
    const float* b_attn = (const float*)d_in[6];
    const float* W_comb = (const float*)d_in[7];
    const float* b_comb = (const float*)d_in[8];
    const float* W_ih   = (const float*)d_in[9];
    const float* b_ih   = (const float*)d_in[10];
    const float* W_hh   = (const float*)d_in[11];
    const float* b_hh   = (const float*)d_in[12];
    const float* W_out  = (const float*)d_in[13];
    const float* b_out  = (const float*)d_in[14];

    float* out_logits = (float*)d_out;                         // (B,784)
    float* out_hnew   = out_logits + (size_t)Bsz * INsz;       // (1,B,H)
    float* out_attnw  = out_hnew + (size_t)Bsz * Hsz;          // (B,L)

    // ---- workspace carve: f16 section then f32 section ----
    _Float16* p16 = (_Float16*)d_ws;
    _Float16* in16    = p16; p16 += (size_t)Bsz * INpad;
    _Float16* h016    = p16; p16 += (size_t)Bsz * Hsz;
    _Float16* Wemb16  = p16; p16 += (size_t)Hsz * INpad;
    _Float16* Wattn16 = p16; p16 += (size_t)Lsz * 2 * Hsz;
    _Float16* Wcomb16 = p16; p16 += (size_t)Hsz * 2 * Hsz;
    _Float16* Wih16   = p16; p16 += (size_t)3 * Hsz * Hsz;
    _Float16* Whh16   = p16; p16 += (size_t)3 * Hsz * Hsz;
    _Float16* Wout16  = p16; p16 += (size_t)INsz * Hsz;
    _Float16* emb16   = p16; p16 += (size_t)Bsz * Hsz;
    _Float16* ap16    = p16; p16 += (size_t)Bsz * Hsz;
    _Float16* x16     = p16; p16 += (size_t)Bsz * Hsz;
    _Float16* hnew16  = p16; p16 += (size_t)Bsz * Hsz;
    float* pf = (float*)p16;
    float* attn_logits = pf; pf += (size_t)Bsz * Lsz;
    float* gi = pf;          pf += (size_t)Bsz * 3 * Hsz;
    float* gh = pf;

    const dim3 blk(256);
    const int MT = Bsz / 128;   // 32

    // ---- one-time fp32 -> f16 conversions (padded where K=784) ----
    cvt16<<<cvtBlocks((long)Bsz * INpad), blk, 0, stream>>>(input,  in16,   INsz, INpad, (long)Bsz * INpad);
    cvt16<<<cvtBlocks((long)Bsz * Hsz),   blk, 0, stream>>>(hidden, h016,   Hsz,  Hsz,   (long)Bsz * Hsz);
    cvt16<<<cvtBlocks((long)Hsz * INpad), blk, 0, stream>>>(W_emb,  Wemb16, INsz, INpad, (long)Hsz * INpad);
    cvt16<<<cvtBlocks((long)Lsz * 2*Hsz), blk, 0, stream>>>(W_attn, Wattn16, 2*Hsz, 2*Hsz, (long)Lsz * 2*Hsz);
    cvt16<<<cvtBlocks((long)Hsz * 2*Hsz), blk, 0, stream>>>(W_comb, Wcomb16, 2*Hsz, 2*Hsz, (long)Hsz * 2*Hsz);
    cvt16<<<cvtBlocks((long)3*Hsz * Hsz), blk, 0, stream>>>(W_ih,   Wih16,  Hsz, Hsz, (long)3*Hsz * Hsz);
    cvt16<<<cvtBlocks((long)3*Hsz * Hsz), blk, 0, stream>>>(W_hh,   Whh16,  Hsz, Hsz, (long)3*Hsz * Hsz);
    cvt16<<<cvtBlocks((long)INsz * Hsz),  blk, 0, stream>>>(W_out,  Wout16, Hsz, Hsz, (long)INsz * Hsz);

    // 1) emb = input @ W_emb^T + b_emb                    (f16 out)
    wmma_gemm<0,0,1><<<dim3(Hsz/128, MT), blk, 0, stream>>>(
        in16, (const _Float16*)nullptr, INpad, INpad, Wemb16, b_emb, (float*)nullptr, emb16, Hsz);
    // 2) attn_logits = [emb|h0] @ W_attn^T + b_attn       (f32 out)
    wmma_gemm<0,1,0><<<dim3(1, MT), blk, 0, stream>>>(
        emb16, h016, Hsz, 2*Hsz, Wattn16, b_attn, attn_logits, (_Float16*)nullptr, Lsz);
    // 3) softmax
    softmax64<<<Bsz, 32, 0, stream>>>(attn_logits, out_attnw);
    // 4) attn_applied
    attn_apply<<<dim3(Bsz, Hsz/256), blk, 0, stream>>>(out_attnw, enc, ap16);
    // 5) x = relu([emb|attn_applied] @ W_comb^T + b_comb) (f16 out)
    wmma_gemm<1,0,1><<<dim3(Hsz/128, MT), blk, 0, stream>>>(
        emb16, ap16, Hsz, 2*Hsz, Wcomb16, b_comb, (float*)nullptr, x16, Hsz);
    // 6) gi = x @ W_ih^T + b_ih                           (f32 out)
    wmma_gemm<0,1,0><<<dim3(3*Hsz/128, MT), blk, 0, stream>>>(
        x16, (const _Float16*)nullptr, Hsz, Hsz, Wih16, b_ih, gi, (_Float16*)nullptr, 3*Hsz);
    // 7) gh = h0 @ W_hh^T + b_hh                          (f32 out)
    wmma_gemm<0,1,0><<<dim3(3*Hsz/128, MT), blk, 0, stream>>>(
        h016, (const _Float16*)nullptr, Hsz, Hsz, Whh16, b_hh, gh, (_Float16*)nullptr, 3*Hsz);
    // 8) GRU gates -> h_new (f32 to d_out, f16 for final GEMM)
    gru_gate<<<(Bsz * Hsz) / 256, blk, 0, stream>>>(gi, gh, hidden, out_hnew, hnew16);
    // 9) logits = h_new @ W_out^T + b_out                 (f32 to d_out)
    wmma_gemm<0,1,0><<<dim3((INsz + 127)/128, MT), blk, 0, stream>>>(
        hnew16, (const _Float16*)nullptr, Hsz, Hsz, Wout16, b_out, out_logits, (_Float16*)nullptr, INsz);
    // 10) log_softmax in place
    logsoftmax784<<<Bsz, blk, 0, stream>>>(out_logits);
}
